// VoxFormer_46024869544006
// MI455X (gfx1250) — matrix operand order repaired and checked by reading, not compiled, not run on hardware
//
#include <hip/hip_runtime.h>
#include <hip/hip_bf16.h>
#include <math.h>

#define DM    128
#define NSAMP 8
#define DFF   512
#define NC    3
#define NSF   2
#define PRED  4
#define BB    2
#define NCAM  5
#define QQ    2500
#define HFD   24
#define WFD   77
#define VXD   32
#define VYD   16
#define VZD   32
#define BQ    (BB * QQ)          // 5000
#define HW    (HFD * WFD)        // 1848
#define CAMS  (BB * NCAM)        // 10
#define TOTM  (CAMS * HW)        // 18480
#define VOXN  (VYD * VXD * VZD)  // 16384

typedef __attribute__((ext_vector_type(16))) _Float16 v16h;
typedef __attribute__((ext_vector_type(8)))  _Float16 v8h;
typedef __attribute__((ext_vector_type(8)))  float    v8f;

#define WMMA_F16(a, b, c) \
  __builtin_amdgcn_wmma_f32_16x16x32_f16(false, (a), false, (b), (short)0, (c), false, false)

// -------- wave32 butterfly reduction --------
__device__ __forceinline__ float wred32(float v) {
#pragma unroll
  for (int m = 16; m > 0; m >>= 1) v += __shfl_xor(v, m, 32);
  return v;
}

// -------- WMMA fragment loaders --------
// A: 16x32 f16 tile from row-major [.. x lda] at (row0, k0); lda multiple of 8,
// k0 multiple of 32 -> both 16B chunks are aligned vector loads.
__device__ __forceinline__ v16h ldA(const _Float16* A, int lda, int row0, int k0) {
  int lane = threadIdx.x & 31;
  int g = lane >> 4, m = lane & 15;
  const _Float16* r = A + (size_t)(row0 + m) * lda + k0 + g * 8;
  v8h lo = *(const v8h*)(r);
  v8h hi = *(const v8h*)(r + 16);
  return __builtin_shufflevector(lo, hi, 0, 1, 2, 3, 4, 5, 6, 7,
                                 8, 9, 10, 11, 12, 13, 14, 15);
}
// B: pre-packed fragment-major weights; lane's 16 halves contiguous (32B aligned)
__device__ __forceinline__ v16h ldBp(const _Float16* P, int frag) {
  int lane = threadIdx.x & 31;
  return *(const v16h*)(P + ((size_t)frag * 32 + lane) * 16);
}

// ============ utility kernels ============
// pack row-major f32 weight [K x N] into WMMA B-fragment order (f16)
__global__ void k_packB(const float* __restrict__ W, _Float16* __restrict__ P, int K, int N) {
  int tid = blockIdx.x * 256 + threadIdx.x;
  if (tid >= K * N) return;
  int h = tid & 15;
  int lane = (tid >> 4) & 31;
  int frag = tid >> 9;
  int ntiles = N >> 4;
  int nt = frag % ntiles, kc = frag / ntiles;
  int n = nt * 16 + (lane & 15);
  int k = kc * 32 + (lane >> 4) * 8 + (h & 7) + (h >> 3) * 16;
  P[tid] = (_Float16)W[(size_t)k * N + n];
}

// image features [cam][c][p] f32 -> pixel-major f16 [cam*HW + p][c]
__global__ void k_img_cvt(const float* __restrict__ src, _Float16* __restrict__ dst) {
  int tid = blockIdx.x * 256 + threadIdx.x;
  if (tid >= TOTM * DM) return;
  int c = tid & 127;
  int rp = tid >> 7;
  int cam = rp / HW, p = rp - cam * HW;
  dst[tid] = (_Float16)src[((size_t)cam * DM + c) * HW + p];
}

__global__ void k_zero32(unsigned int* __restrict__ p, int n) {
  int i = blockIdx.x * 256 + threadIdx.x;
  if (i < n) p[i] = 0u;
}

// ============ LayerNorm + sampling-offset + attention head (wave per row) ============
__global__ void k_ln_dp_att(const float* __restrict__ q, const float* __restrict__ gg,
                            const float* __restrict__ bb_, const float* __restrict__ Wdp,
                            const float* __restrict__ bdp, int DPO,
                            const float* __restrict__ Wa, const float* __restrict__ ba,
                            float* __restrict__ dpo, float* __restrict__ atto) {
  int wid = threadIdx.x >> 5, lane = threadIdx.x & 31;
  int row = blockIdx.x * 8 + wid;
  if (row >= BQ) return;
  int c = lane * 4;
  const float4 xv = *(const float4*)(q + (size_t)row * DM + c);
  float xs[4] = {xv.x, xv.y, xv.z, xv.w};
  float mean = wred32(xs[0] + xs[1] + xs[2] + xs[3]) * (1.0f / DM);
  float vs = 0.f;
#pragma unroll
  for (int j = 0; j < 4; ++j) { float d = xs[j] - mean; vs += d * d; }
  float rstd = rsqrtf(wred32(vs) * (1.0f / DM) + 1e-5f);
  float ln[4];
#pragma unroll
  for (int j = 0; j < 4; ++j) ln[j] = (xs[j] - mean) * rstd * gg[c + j] + bb_[c + j];
  for (int o = 0; o < DPO; ++o) {
    float p = 0.f;
#pragma unroll
    for (int j = 0; j < 4; ++j) p += ln[j] * Wdp[(size_t)(c + j) * DPO + o];
    p = wred32(p);
    if (lane == 0) dpo[(size_t)row * DPO + o] = p + bdp[o];
  }
  float lg[NSAMP];
#pragma unroll
  for (int o = 0; o < NSAMP; ++o) {
    float p = 0.f;
#pragma unroll
    for (int j = 0; j < 4; ++j) p += ln[j] * Wa[(size_t)(c + j) * NSAMP + o];
    lg[o] = wred32(p) + ba[o];
  }
  float mx = lg[0];
#pragma unroll
  for (int o = 1; o < NSAMP; ++o) mx = fmaxf(mx, lg[o]);
  float se = 0.f;
#pragma unroll
  for (int o = 0; o < NSAMP; ++o) { lg[o] = __expf(lg[o] - mx); se += lg[o]; }
  float inv = 1.0f / se;
  if (lane < NSAMP) atto[(size_t)row * NSAMP + lane] = lg[lane] * inv;
}

// ============ LayerNorm -> f16 buffer (wave per row) ============
__global__ void k_ln16(const float* __restrict__ q, const float* __restrict__ gg,
                       const float* __restrict__ bb_, _Float16* __restrict__ out) {
  int wid = threadIdx.x >> 5, lane = threadIdx.x & 31;
  int row = blockIdx.x * 8 + wid;
  if (row >= BQ) return;
  int c = lane * 4;
  const float4 xv = *(const float4*)(q + (size_t)row * DM + c);
  float xs[4] = {xv.x, xv.y, xv.z, xv.w};
  float mean = wred32(xs[0] + xs[1] + xs[2] + xs[3]) * (1.0f / DM);
  float vs = 0.f;
#pragma unroll
  for (int j = 0; j < 4; ++j) { float d = xs[j] - mean; vs += d * d; }
  float rstd = rsqrtf(wred32(vs) * (1.0f / DM) + 1e-5f);
  v8h o;
#pragma unroll
  for (int j = 0; j < 4; ++j)
    o[j] = (_Float16)((xs[j] - mean) * rstd * gg[c + j] + bb_[c + j]);
  // store 4 halves (8B) contiguously
  *(uint2*)(out + (size_t)row * DM + c) = *(uint2*)&o;
}

// ============ value projection GEMM: imgT[18480 x 128] x W[128 x 128] -> vfT (WMMA) ============
__global__ void k_vproj(const _Float16* __restrict__ imgT, const _Float16* __restrict__ Pw,
                        const float* __restrict__ bias, _Float16* __restrict__ vfT) {
  __builtin_prefetch(Pw, 0, 1);
  int m0 = blockIdx.x * 16;
  int wave = threadIdx.x >> 5, lane = threadIdx.x & 31;
  int g = lane >> 4, nl = lane & 15;
  int n0 = wave * 16;
  v8f acc = {};
#pragma unroll
  for (int kc = 0; kc < 4; ++kc)
    acc = WMMA_F16(ldA(imgT, DM, m0, kc * 32), ldBp(Pw, kc * 8 + wave), acc);
  float bn = bias[n0 + nl];
#pragma unroll
  for (int rr = 0; rr < 8; ++rr) {
    int m = rr + g * 8;
    vfT[(size_t)(m0 + m) * DM + n0 + nl] = (_Float16)(acc[rr] + bn);
  }
}

// ============ cross-attn bilinear gather + att/mask fold -> agg[B*Q,128] ============
__global__ void k_gather2d(const _Float16* __restrict__ vfT, const float* __restrict__ ref,
                           const float* __restrict__ dp, const float* __restrict__ att,
                           const int* __restrict__ mask, _Float16* __restrict__ aggh,
                           float* __restrict__ fb) {
  int row = blockIdx.x;
  int b = row / QQ, qq = row - b * QQ;
  int c = threadIdx.x;
  float at[NSAMP], dx[NSAMP], dy[NSAMP];
#pragma unroll
  for (int s = 0; s < NSAMP; ++s) {
    at[s] = att[(size_t)row * NSAMP + s];
    dx[s] = dp[(size_t)row * 16 + s * 2 + 0];
    dy[s] = dp[(size_t)row * 16 + s * 2 + 1];
  }
  float acc = 0.f, summf = 0.f;
  for (int n = 0; n < NCAM; ++n) {
    int mv = mask[((size_t)b * NCAM + n) * QQ + qq];
    if (!mv) continue;
    summf += 1.0f;
    float rx = ref[(((size_t)b * NCAM + n) * QQ + qq) * 2 + 0];
    float ry = ref[(((size_t)b * NCAM + n) * QQ + qq) * 2 + 1];
    const _Float16* f = vfT + (size_t)(b * NCAM + n) * HW * DM;
#pragma unroll
    for (int s = 0; s < NSAMP; ++s) {
      // pd=(ref+dp)*2-1; x=(pd+1)/2*(W-1) == (ref+dp)*(W-1)
      float x = (rx + dx[s]) * (WFD - 1);
      float y = (ry + dy[s]) * (HFD - 1);
      float xf = floorf(x), yf = floorf(y);
      int x0 = (int)xf, y0 = (int)yf;
      float xd = x - xf, yd = y - yf;
      float v = 0.f;
#pragma unroll
      for (int cy = 0; cy < 2; ++cy)
#pragma unroll
        for (int cx = 0; cx < 2; ++cx) {
          int xi = x0 + cx, yi = y0 + cy;
          if (xi >= 0 && xi < WFD && yi >= 0 && yi < HFD) {
            float wgt = (cx ? xd : 1.0f - xd) * (cy ? yd : 1.0f - yd);
            v += wgt * (float)f[(size_t)(yi * WFD + xi) * DM + c];  // coalesced over c
          }
        }
      acc += at[s] * v;
    }
  }
  float denom = summf + 1e-6f;
  aggh[(size_t)row * DM + c] = (_Float16)(acc / denom);
  if (c == 0) fb[row] = summf / denom;
}

// ============ cross: s = agg@Ww + fb*wb ; out = s@Wop + opb ; q += out ============
__global__ void k_cross_out(const _Float16* __restrict__ aggh, const float* __restrict__ fb,
                            const _Float16* __restrict__ Pww, const float* __restrict__ wb,
                            const _Float16* __restrict__ Pop, const float* __restrict__ opb,
                            float* __restrict__ q) {
  __shared__ __align__(16) _Float16 sA[16 * DM];
  __shared__ __align__(16) _Float16 sS[16 * DM];
  __shared__ float sFb[16];
  __builtin_prefetch(Pww, 0, 1);
  int m0 = blockIdx.x * 16;
  {
    int i = threadIdx.x;              // 256 chunks of 8 halves = 16x128
    int r = i >> 4, ch = i & 15;
    int row = m0 + r; if (row > BQ - 1) row = BQ - 1;
    *(v8h*)(sA + r * DM + ch * 8) = *(const v8h*)(aggh + (size_t)row * DM + ch * 8);
  }
  if (threadIdx.x < 16) {
    int row = m0 + threadIdx.x; if (row > BQ - 1) row = BQ - 1;
    sFb[threadIdx.x] = fb[row];
  }
  __syncthreads();
  int wave = threadIdx.x >> 5, lane = threadIdx.x & 31;
  int g = lane >> 4, nl = lane & 15;
  int n0 = wave * 16;
  v8f acc = {};
#pragma unroll
  for (int kc = 0; kc < 4; ++kc)
    acc = WMMA_F16(ldA(sA, DM, 0, kc * 32), ldBp(Pww, kc * 8 + wave), acc);
  float wbn = wb[n0 + nl];
#pragma unroll
  for (int rr = 0; rr < 8; ++rr) {
    int m = rr + g * 8;
    sS[m * DM + n0 + nl] = (_Float16)(acc[rr] + sFb[m] * wbn);
  }
  __syncthreads();
  v8f acc2 = {};
#pragma unroll
  for (int kc = 0; kc < 4; ++kc)
    acc2 = WMMA_F16(ldA(sS, DM, 0, kc * 32), ldBp(Pop, kc * 8 + wave), acc2);
  float opn = opb[n0 + nl];
#pragma unroll
  for (int rr = 0; rr < 8; ++rr) {
    int m = rr + g * 8, row = m0 + m;
    if (row < BQ) q[(size_t)row * DM + n0 + nl] += acc2[rr] + opn;
  }
}

// ============ FFN: h1=gelu(ln@W1+b1) ; q += h1@W2+b2 (WMMA, LDS-staged) ============
__global__ void k_ffn(const _Float16* __restrict__ lnh, const _Float16* __restrict__ P1,
                      const float* __restrict__ b1, const _Float16* __restrict__ P2,
                      const float* __restrict__ b2, float* __restrict__ q) {
  __shared__ __align__(16) _Float16 sX[16 * DM];
  __shared__ __align__(16) _Float16 sH[16 * DFF];
  __builtin_prefetch(P1, 0, 1);
  int m0 = blockIdx.x * 16;
  {
    int i = threadIdx.x;
    int r = i >> 4, ch = i & 15;
    int row = m0 + r; if (row > BQ - 1) row = BQ - 1;
    *(v8h*)(sX + r * DM + ch * 8) = *(const v8h*)(lnh + (size_t)row * DM + ch * 8);
  }
  __syncthreads();
  int wave = threadIdx.x >> 5, lane = threadIdx.x & 31;
  int g = lane >> 4, nl = lane & 15;
#pragma unroll
  for (int t = 0; t < 4; ++t) {
    int nt = wave + t * 8;
    int n0 = nt * 16;
    v8f acc = {};
#pragma unroll
    for (int kc = 0; kc < 4; ++kc)
      acc = WMMA_F16(ldA(sX, DM, 0, kc * 32), ldBp(P1, kc * 32 + nt), acc);
    float b1n = b1[n0 + nl];
#pragma unroll
    for (int rr = 0; rr < 8; ++rr) {
      int m = rr + g * 8;
      float x = acc[rr] + b1n;
      float ge = 0.5f * x * (1.0f + erff(x * 0.70710678118654752f));
      sH[m * DFF + n0 + nl] = (_Float16)ge;
    }
  }
  __syncthreads();
  int n0 = wave * 16;
  v8f acc2 = {};
#pragma unroll
  for (int kc = 0; kc < 16; ++kc)
    acc2 = WMMA_F16(ldA(sH, DFF, 0, kc * 32), ldBp(P2, kc * 8 + wave), acc2);
  float b2n = b2[n0 + nl];
#pragma unroll
  for (int rr = 0; rr < 8; ++rr) {
    int m = rr + g * 8, row = m0 + m;
    if (row < BQ) q[(size_t)row * DM + n0 + nl] += acc2[rr] + b2n;
  }
}

// ============ voxel scatter (channel-contiguous volume) ============
__global__ void k_scatter(const float* __restrict__ q, const int* __restrict__ xi,
                          const int* __restrict__ yi, const int* __restrict__ zi,
                          _Float16* __restrict__ vol) {
  int tid = blockIdx.x * 256 + threadIdx.x;
  if (tid >= BQ * DM) return;
  int row = tid >> 7, c = tid & 127;
  int b = row / QQ;
  int sp = yi[row] * (VXD * VZD) + xi[row] * VZD + zi[row];
  vol[((size_t)b * VOXN + sp) * DM + c] = (_Float16)q[(size_t)row * DM + c];
}

__global__ void k_many(const int* __restrict__ mask, float* __restrict__ many) {
  int row = blockIdx.x * 256 + threadIdx.x;
  if (row >= BQ) return;
  int b = row / QQ, qq = row - b * QQ;
  int any = 0;
  for (int n = 0; n < NCAM; ++n) any |= (mask[((size_t)b * NCAM + n) * QQ + qq] != 0);
  many[row] = any ? 1.0f : 0.0f;
}

// ============ self-attn trilinear gather + att fold -> agg ============
__global__ void k_gather3d(const _Float16* __restrict__ vol, const float* __restrict__ ref3,
                           const float* __restrict__ dp, const float* __restrict__ att,
                           _Float16* __restrict__ aggh) {
  int row = blockIdx.x;
  int b = row / QQ;
  int c = threadIdx.x;
  const _Float16* f = vol + (size_t)b * VOXN * DM;
  float r0 = ref3[(size_t)row * 3 + 0], r1 = ref3[(size_t)row * 3 + 1], r2 = ref3[(size_t)row * 3 + 2];
  float acc = 0.f;
#pragma unroll
  for (int s = 0; s < NSAMP; ++s) {
    float p0 = r0 + dp[(size_t)row * 24 + s * 3 + 0];
    float p1 = r1 + dp[(size_t)row * 24 + s * 3 + 1];
    float p2 = r2 + dp[(size_t)row * 24 + s * 3 + 2];
    // grid channels after [1,0,2] permute: x<-p1 (W=VZ), y<-p0 (H=VX), z<-p2 (D=VY)
    float x = (p1 + 1.0f) * 0.5f * (VZD - 1);
    float y = (p0 + 1.0f) * 0.5f * (VXD - 1);
    float z = (p2 + 1.0f) * 0.5f * (VYD - 1);
    float xf = floorf(x), yf = floorf(y), zf = floorf(z);
    int x0 = (int)xf, y0 = (int)yf, z0 = (int)zf;
    float xd = x - xf, yd = y - yf, zd = z - zf;
    float v = 0.f;
#pragma unroll
    for (int cz = 0; cz < 2; ++cz)
#pragma unroll
      for (int cy = 0; cy < 2; ++cy)
#pragma unroll
        for (int cx = 0; cx < 2; ++cx) {
          int xi = x0 + cx, yi = y0 + cy, zi = z0 + cz;
          if (xi >= 0 && xi < VZD && yi >= 0 && yi < VXD && zi >= 0 && zi < VYD) {
            float wgt = (cx ? xd : 1.f - xd) * (cy ? yd : 1.f - yd) * (cz ? zd : 1.f - zd);
            v += wgt * (float)f[(size_t)(zi * (VXD * VZD) + yi * VZD + xi) * DM + c];
          }
        }
    acc += att[(size_t)row * NSAMP + s] * v;
  }
  aggh[(size_t)row * DM + c] = (_Float16)acc;
}

// ============ self: v=agg@Vp+vb ; wf=(v@Ww+wb)*m_any ; q += wf@Wop+opb ============
__global__ void k_self_out(const _Float16* __restrict__ aggh, const float* __restrict__ many,
                           const _Float16* __restrict__ Pvp, const float* __restrict__ vpb,
                           const _Float16* __restrict__ Pww, const float* __restrict__ wb,
                           const _Float16* __restrict__ Pop, const float* __restrict__ opb,
                           float* __restrict__ q) {
  __shared__ __align__(16) _Float16 sA[16 * DM];
  __shared__ __align__(16) _Float16 sB2[16 * DM];
  __shared__ float sM[16];
  __builtin_prefetch(Pvp, 0, 1);
  int m0 = blockIdx.x * 16;
  {
    int i = threadIdx.x;
    int r = i >> 4, ch = i & 15;
    int row = m0 + r; if (row > BQ - 1) row = BQ - 1;
    *(v8h*)(sA + r * DM + ch * 8) = *(const v8h*)(aggh + (size_t)row * DM + ch * 8);
  }
  if (threadIdx.x < 16) {
    int row = m0 + threadIdx.x; if (row > BQ - 1) row = BQ - 1;
    sM[threadIdx.x] = many[row];
  }
  __syncthreads();
  int wave = threadIdx.x >> 5, lane = threadIdx.x & 31;
  int g = lane >> 4, nl = lane & 15;
  int n0 = wave * 16;
  v8f acc = {};
#pragma unroll
  for (int kc = 0; kc < 4; ++kc)
    acc = WMMA_F16(ldA(sA, DM, 0, kc * 32), ldBp(Pvp, kc * 8 + wave), acc);
  float vpn = vpb[n0 + nl];
#pragma unroll
  for (int rr = 0; rr < 8; ++rr) {
    int m = rr + g * 8;
    sB2[m * DM + n0 + nl] = (_Float16)(acc[rr] + vpn);
  }
  __syncthreads();
  v8f acc2 = {};
#pragma unroll
  for (int kc = 0; kc < 4; ++kc)
    acc2 = WMMA_F16(ldA(sB2, DM, 0, kc * 32), ldBp(Pww, kc * 8 + wave), acc2);
  __syncthreads();
  float wbn = wb[n0 + nl];
#pragma unroll
  for (int rr = 0; rr < 8; ++rr) {
    int m = rr + g * 8;
    sA[m * DM + n0 + nl] = (_Float16)((acc2[rr] + wbn) * sM[m]);
  }
  __syncthreads();
  v8f acc3 = {};
#pragma unroll
  for (int kc = 0; kc < 4; ++kc)
    acc3 = WMMA_F16(ldA(sA, DM, 0, kc * 32), ldBp(Pop, kc * 8 + wave), acc3);
  float opn = opb[n0 + nl];
#pragma unroll
  for (int rr = 0; rr < 8; ++rr) {
    int m = rr + g * 8, row = m0 + m;
    if (row < BQ) q[(size_t)row * DM + n0 + nl] += acc3[rr] + opn;
  }
}

// ============ classifier head (wave per row) ============
__global__ void k_logits(const float* __restrict__ q, const float* __restrict__ Wc,
                         const float* __restrict__ bc, float* __restrict__ out) {
  int wid = threadIdx.x >> 5, lane = threadIdx.x & 31;
  int row = blockIdx.x * 8 + wid;
  if (row >= BQ) return;
  int c = lane * 4;
  const float4 xv = *(const float4*)(q + (size_t)row * DM + c);
  float xs[4] = {xv.x, xv.y, xv.z, xv.w};
#pragma unroll
  for (int p = 0; p < PRED; ++p) {
    float s = 0.f;
#pragma unroll
    for (int j = 0; j < 4; ++j) s += xs[j] * Wc[(size_t)(c + j) * PRED + p];
    s = wred32(s);
    if (lane == 0) out[(size_t)row * PRED + p] = s + bc[p];
  }
}

// ===================================================================
extern "C" void kernel_launch(void* const* d_in, const int* in_sizes, int n_in,
                              void* d_out, int out_size, void* d_ws, size_t ws_size,
                              hipStream_t stream) {
  (void)in_sizes; (void)n_in; (void)out_size; (void)ws_size;
  const float* in_query = (const float*)d_in[0];
  const float* ref_points = (const float*)d_in[1];
  const float* ref3 = (const float*)d_in[2];
  const float* imgf = (const float*)d_in[3];
  const int* mask = (const int*)d_in[4];
  const int* x_idx = (const int*)d_in[5];
  const int* y_idx = (const int*)d_in[6];
  const int* z_idx = (const int*)d_in[7];
  const float* c_ln1_g = (const float*)d_in[8];
  const float* c_ln1_b = (const float*)d_in[9];
  const float* c_dp_w = (const float*)d_in[10];
  const float* c_dp_b = (const float*)d_in[11];
  const float* c_a_w = (const float*)d_in[12];
  const float* c_a_b = (const float*)d_in[13];
  const float* c_w_w = (const float*)d_in[14];
  const float* c_w_b = (const float*)d_in[15];
  const float* c_vp_w = (const float*)d_in[16];
  const float* c_vp_b = (const float*)d_in[17];
  const float* c_op_w = (const float*)d_in[18];
  const float* c_op_b = (const float*)d_in[19];
  const float* c_ln2_g = (const float*)d_in[20];
  const float* c_ln2_b = (const float*)d_in[21];
  const float* c_ff1_w = (const float*)d_in[22];
  const float* c_ff1_b = (const float*)d_in[23];
  const float* c_ff2_w = (const float*)d_in[24];
  const float* c_ff2_b = (const float*)d_in[25];
  const float* s_ln1_g = (const float*)d_in[26];
  const float* s_ln1_b = (const float*)d_in[27];
  const float* s_dp_w = (const float*)d_in[28];
  const float* s_dp_b = (const float*)d_in[29];
  const float* s_a_w = (const float*)d_in[30];
  const float* s_a_b = (const float*)d_in[31];
  const float* s_w_w = (const float*)d_in[32];
  const float* s_w_b = (const float*)d_in[33];
  const float* s_vp_w = (const float*)d_in[34];
  const float* s_vp_b = (const float*)d_in[35];
  const float* s_op_w = (const float*)d_in[36];
  const float* s_op_b = (const float*)d_in[37];
  const float* s_ln2_g = (const float*)d_in[38];
  const float* s_ln2_b = (const float*)d_in[39];
  const float* s_ff1_w = (const float*)d_in[40];
  const float* s_ff1_b = (const float*)d_in[41];
  const float* s_ff2_w = (const float*)d_in[42];
  const float* s_ff2_b = (const float*)d_in[43];
  const float* cls_w = (const float*)d_in[44];
  const float* cls_b = (const float*)d_in[45];

  // ---- workspace layout ----
  char* base = (char*)d_ws;
  size_t off = 0;
  auto A = [&](size_t bytes) -> void* {
    void* p = base + off;
    off = (off + bytes + 255) & ~(size_t)255;
    return p;
  };
  float* qw = (float*)A(sizeof(float) * BQ * DM);
  _Float16* aggh = (_Float16*)A(sizeof(_Float16) * BQ * DM);
  _Float16* lnh = (_Float16*)A(sizeof(_Float16) * BQ * DM);
  float* dpb = (float*)A(sizeof(float) * BQ * 24);
  float* attb = (float*)A(sizeof(float) * BQ * NSAMP);
  float* fbb = (float*)A(sizeof(float) * BQ);
  float* manyb = (float*)A(sizeof(float) * BQ);
  _Float16 *cw_vp[NC], *cw_w[NC], *cw_op[NC], *cw_f1[NC], *cw_f2[NC];
  for (int i = 0; i < NC; ++i) {
    cw_vp[i] = (_Float16*)A(2u * DM * DM);
    cw_w[i]  = (_Float16*)A(2u * DM * DM);
    cw_op[i] = (_Float16*)A(2u * DM * DM);
    cw_f1[i] = (_Float16*)A(2u * DM * DFF);
    cw_f2[i] = (_Float16*)A(2u * DFF * DM);
  }
  _Float16 *sw_vp[NSF], *sw_w[NSF], *sw_op[NSF], *sw_f1[NSF], *sw_f2[NSF];
  for (int i = 0; i < NSF; ++i) {
    sw_vp[i] = (_Float16*)A(2u * DM * DM);
    sw_w[i]  = (_Float16*)A(2u * DM * DM);
    sw_op[i] = (_Float16*)A(2u * DM * DM);
    sw_f1[i] = (_Float16*)A(2u * DM * DFF);
    sw_f2[i] = (_Float16*)A(2u * DFF * DM);
  }
  // big region: imgT + vfT (cross phase) overlapped with voxel volume (self phase)
  size_t imgBytes = 2ull * TOTM * DM;   // 4.73 MB
  char* region = (char*)A(imgBytes * 2);  // 9.46 MB >= vol (8.39 MB)
  _Float16* imgT = (_Float16*)region;
  _Float16* vfT = (_Float16*)(region + imgBytes);
  _Float16* vol = (_Float16*)region;

  // ---- prep ----
  hipMemcpyAsync(qw, in_query, sizeof(float) * BQ * DM, hipMemcpyDeviceToDevice, stream);
  auto pack = [&](const float* s, _Float16* d, int K, int N) {
    k_packB<<<(K * N + 255) / 256, 256, 0, stream>>>(s, d, K, N);
  };
  k_img_cvt<<<(TOTM * DM + 255) / 256, 256, 0, stream>>>(imgf, imgT);
  for (int i = 0; i < NC; ++i) {
    pack(c_vp_w + (size_t)i * DM * DM, cw_vp[i], DM, DM);
    pack(c_w_w + (size_t)i * DM * DM, cw_w[i], DM, DM);
    pack(c_op_w + (size_t)i * DM * DM, cw_op[i], DM, DM);
    pack(c_ff1_w + (size_t)i * DM * DFF, cw_f1[i], DM, DFF);
    pack(c_ff2_w + (size_t)i * DFF * DM, cw_f2[i], DFF, DM);
  }
  for (int i = 0; i < NSF; ++i) {
    pack(s_vp_w + (size_t)i * DM * DM, sw_vp[i], DM, DM);
    pack(s_w_w + (size_t)i * DM * DM, sw_w[i], DM, DM);
    pack(s_op_w + (size_t)i * DM * DM, sw_op[i], DM, DM);
    pack(s_ff1_w + (size_t)i * DM * DFF, sw_f1[i], DM, DFF);
    pack(s_ff2_w + (size_t)i * DFF * DM, sw_f2[i], DFF, DM);
  }
  k_many<<<(BQ + 255) / 256, 256, 0, stream>>>(mask, manyb);

  // ---- cross-attention stack ----
  for (int i = 0; i < NC; ++i) {
    k_ln_dp_att<<<(BQ + 7) / 8, 256, 0, stream>>>(
        qw, c_ln1_g + i * DM, c_ln1_b + i * DM,
        c_dp_w + (size_t)i * DM * NSAMP * 2, c_dp_b + i * NSAMP * 2, NSAMP * 2,
        c_a_w + (size_t)i * DM * NSAMP, c_a_b + i * NSAMP, dpb, attb);
    k_vproj<<<TOTM / 16, 256, 0, stream>>>(imgT, cw_vp[i], c_vp_b + i * DM, vfT);
    k_gather2d<<<BQ, DM, 0, stream>>>(vfT, ref_points, dpb, attb, mask, aggh, fbb);
    k_cross_out<<<(BQ + 15) / 16, 256, 0, stream>>>(aggh, fbb, cw_w[i], c_w_b + i * DM,
                                                    cw_op[i], c_op_b + i * DM, qw);
    k_ln16<<<(BQ + 7) / 8, 256, 0, stream>>>(qw, c_ln2_g + i * DM, c_ln2_b + i * DM, lnh);
    k_ffn<<<(BQ + 15) / 16, 256, 0, stream>>>(lnh, cw_f1[i], c_ff1_b + i * DFF,
                                              cw_f2[i], c_ff2_b + i * DM, qw);
  }

  // ---- self-attention stack ----
  for (int i = 0; i < NSF; ++i) {
    int volWords = BB * DM * VOXN / 2;
    k_zero32<<<(volWords + 255) / 256, 256, 0, stream>>>((unsigned int*)vol, volWords);
    k_scatter<<<(BQ * DM + 255) / 256, 256, 0, stream>>>(qw, x_idx, y_idx, z_idx, vol);
    k_ln_dp_att<<<(BQ + 7) / 8, 256, 0, stream>>>(
        qw, s_ln1_g + i * DM, s_ln1_b + i * DM,
        s_dp_w + (size_t)i * DM * NSAMP * 3, s_dp_b + i * NSAMP * 3, NSAMP * 3,
        s_a_w + (size_t)i * DM * NSAMP, s_a_b + i * NSAMP, dpb, attb);
    k_gather3d<<<BQ, DM, 0, stream>>>(vol, ref3, dpb, attb, aggh);
    k_self_out<<<(BQ + 15) / 16, 256, 0, stream>>>(aggh, manyb, sw_vp[i], s_vp_b + i * DM,
                                                   sw_w[i], s_w_b + i * DM,
                                                   sw_op[i], s_op_b + i * DM, qw);
    k_ln16<<<(BQ + 7) / 8, 256, 0, stream>>>(qw, s_ln2_g + i * DM, s_ln2_b + i * DM, lnh);
    k_ffn<<<(BQ + 15) / 16, 256, 0, stream>>>(lnh, sw_f1[i], s_ff1_b + i * DFF,
                                              sw_f2[i], s_ff2_b + i * DM, qw);
  }

  k_logits<<<(BQ + 7) / 8, 256, 0, stream>>>(qw, cls_w, cls_b, (float*)d_out);
}
